// Model_15341623181773
// MI455X (gfx1250) — matrix operand order repaired and checked by reading, not compiled, not run on hardware
//
#include <hip/hip_runtime.h>
#include <hip/hip_bf16.h>

typedef __attribute__((ext_vector_type(16))) _Float16 v16h;
typedef __attribute__((ext_vector_type(8)))  float    v8f;

#define Bn   512
#define Tn   49
#define Vn   34
#define Fn   16
#define Hn   64
#define WINn 4
#define Wn   46
#define NIn  (Bn*Wn)        /* 23552 windows */
#define H3   (3*Hn)         /* 192 */

/* ---------------- workspace layout (float offsets) ---------------- */
#define OFF_PEWIN  0
#define OFF_PEV    64
#define OFF_PEW    608
#define OFF_CMAX   3552
#define OFF_CUM    3584
#define OFF_LAST   (OFF_CUM  + Bn*Tn*Vn)
#define OFF_MEAN   (OFF_LAST + Bn*Tn*Vn)
#define OFF_XIMP   (OFF_MEAN + Bn*Vn)
#define OFF_SOUT   (OFF_XIMP + Bn*Tn*Vn)
#define OFF_TVAL   (OFF_SOUT + NIn*Fn)
#define OFF_A      (OFF_TVAL + Bn*Hn)         /* region A: enc -> message -> Qt/Kt/Vt */
#define SZ_A       (Bn*Tn*Vn*Fn)
#define OFF_B      (OFF_A + SZ_A)             /* region B: att_out -> xW0|g1|xW1|g2 */

__device__ __forceinline__ v8f wmma16(v16h a, v16h b, v8f c) {
  return __builtin_amdgcn_wmma_f32_16x16x32_f16(false, a, false, b, (short)0, c, false, false);
}
__device__ __forceinline__ int a_kmap(int e, int half) {
  return half * 16 + e;
}
__device__ __forceinline__ float pe_val(int pos, int c, int d) {
  float div = expf(-(float)(2 * (c >> 1)) * (logf(10000.0f) / (float)d));
  float x = (float)pos * div;
  return (c & 1) ? cosf(x) : sinf(x);
}

/* ---------------- K0: PE tables + reset max ---------------- */
__global__ void k_init(float* pe_win, float* pe_v, float* pe_w, unsigned* cmax) {
  int tid = blockIdx.x * blockDim.x + threadIdx.x;
  if (tid == 0) *cmax = 0u;
  if (tid < WINn * Fn) pe_win[tid] = pe_val(tid / Fn, tid % Fn, Fn);
  if (tid < Vn * Fn)   pe_v[tid]   = pe_val(tid / Fn, tid % Fn, Fn);
  if (tid < Wn * Hn)   pe_w[tid]   = pe_val(tid / Hn, tid % Hn, Hn);
}

/* ---------------- K1: masked cum-delta + LOCF scan ---------------- */
__global__ void k_scan(const float* __restrict__ arr, const float* __restrict__ mask,
                       const float* __restrict__ tim,
                       float* __restrict__ cum, float* __restrict__ last,
                       float* __restrict__ meanv, unsigned* __restrict__ cmax) {
  int id = blockIdx.x * blockDim.x + threadIdx.x;
  if (id >= Bn * Vn) return;
  int b = id / Vn, v = id % Vn;
  float x = arr[(b * Tn) * Vn + v];
  float c = 0.0f, msum = x, lmax = 0.0f;
  float tprev = tim[0 * Bn + b];
  cum[(b * Tn) * Vn + v] = 0.0f;
  last[(b * Tn) * Vn + v] = x;
  for (int t = 1; t < Tn; ++t) {
    float tt = tim[t * Bn + b];
    float d = tt - tprev; tprev = tt;
    float a  = arr[(b * Tn + t) * Vn + v];
    float mk = mask[(b * Tn + t) * Vn + v];
    float missing = (mk == 0.0f) ? 1.0f : 0.0f;
    c = d + c * missing;                 /* resets at observations */
    x = (missing > 0.0f) ? x : a;        /* LOCF */
    cum[(b * Tn + t) * Vn + v] = c;
    last[(b * Tn + t) * Vn + v] = x;
    lmax = fmaxf(lmax, c);
    msum += a;
  }
  meanv[b * Vn + v] = msum / (float)Tn;
  atomicMax(cmax, __float_as_uint(lmax));  /* cum >= 0 -> uint order == float order */
}

/* ---------------- K2: impute ---------------- */
__global__ void k_apply(const float* __restrict__ arr, const float* __restrict__ mask,
                        const float* __restrict__ cum, const float* __restrict__ last,
                        const float* __restrict__ meanv, const unsigned* __restrict__ cmax,
                        const float* __restrict__ wdec, const float* __restrict__ bdec,
                        float* __restrict__ ximp) {
  int id = blockIdx.x * blockDim.x + threadIdx.x;
  if (id >= Bn * Tn * Vn) return;
  int v = id % Vn;
  int b = id / (Tn * Vn);
  float cm = __uint_as_float(*cmax);
  float cn = cum[id] / cm;
  float e = fminf(fmaxf(wdec[v] * cn + bdec[v], 0.0f), 1000.0f);
  float dec = expf(-e);
  float mk = mask[id];
  float mean = meanv[b * Vn + v];
  ximp[id] = arr[id] * mk + (1.0f - mk) * (dec * last[id] + (1.0f - dec) * mean);
}

/* ---------------- K3: per-variable encoder MLP ---------------- */
__global__ void k_enc(const float* __restrict__ ximp, const float* __restrict__ mask,
                      const float* __restrict__ W1, const float* __restrict__ b1,
                      const float* __restrict__ W2, const float* __restrict__ b2,
                      float* __restrict__ enc) {
  int id = blockIdx.x * blockDim.x + threadIdx.x;
  if (id >= Bn * Tn * Vn) return;
  int v = id % Vn;
  float x = ximp[id], mk = mask[id];
  float h[Fn];
#pragma unroll
  for (int j = 0; j < Fn; ++j) {
    float a = x * W1[v * 2 * Fn + j] + mk * W1[v * 2 * Fn + Fn + j] + b1[v * Fn + j];
    h[j] = fmaxf(a, 0.0f);
  }
#pragma unroll
  for (int g = 0; g < Fn; ++g) {
    float s = b2[v * Fn + g];
#pragma unroll
    for (int j = 0; j < Fn; ++j) s += h[j] * W2[(v * Fn + j) * Fn + g];
    enc[id * Fn + g] = s;
  }
}

/* ---------------- K4: time-lag attention (WMMA, wave = 1 var x 4 windows) ------- */
__global__ void k_tla(const float* __restrict__ enc, const float* __restrict__ pe_win,
                      const float* __restrict__ Wq, const float* __restrict__ bq,
                      const float* __restrict__ Wk, const float* __restrict__ bk,
                      const float* __restrict__ Wv, const float* __restrict__ bv,
                      float* __restrict__ att_out) {
  __shared__ float sQ[8][256];
  __shared__ float sK[8][256];
  __shared__ float sV[8][256];
  __shared__ float sC[8][256];
  int lane = threadIdx.x & 31;
  int wib  = threadIdx.x >> 5;
  int wid  = blockIdx.x * 8 + wib;          /* grid exact: 34*5888 waves */
  int v  = wid % Vn;
  int ib = wid / Vn;
  int i0 = ib * 4;
  int half = lane >> 4, lr = lane & 15;
  int il = lr >> 2, l = lr & 3;             /* tile row = il*4 + l */
  int i  = i0 + il;
  int bb = i / Wn, w = i % Wn;

  v16h a;
#pragma unroll
  for (int e = 0; e < 16; ++e) {
    int k = a_kmap(e, half);
    float val = 0.0f;
    if (k < Fn) {
      int t = w + l;
      val = enc[(((bb * Tn + t) * Vn) + v) * Fn + k] + pe_win[l * Fn + k];
    }
    a[e] = (_Float16)val;
  }
  v16h fq, fk, fv;
#pragma unroll
  for (int e = 0; e < 16; ++e) {
    int k = half * 16 + e;
    float wq = 0.0f, wk = 0.0f, wv = 0.0f;
    if (k < Fn) {
      wq = Wq[(v * Fn + k) * Fn + lr];
      wk = Wk[(v * Fn + k) * Fn + lr];
      wv = Wv[(v * Fn + k) * Fn + lr];
    }
    fq[e] = (_Float16)wq; fk[e] = (_Float16)wk; fv[e] = (_Float16)wv;
  }
  v8f z = {};
  v8f cq = wmma16(a, fq, z);
  v8f ck = wmma16(a, fk, z);
  v8f cv = wmma16(a, fv, z);
#pragma unroll
  for (int j = 0; j < 8; ++j) {
    int r = j + 8 * half;
    sQ[wib][r * 16 + lr] = cq[j] + bq[v * Fn + lr];
    sK[wib][r * 16 + lr] = ck[j] + bk[v * Fn + lr];
    sV[wib][r * 16 + lr] = cv[j] + bv[v * Fn + lr];
  }
  __syncthreads();
  /* per-(window,lag) 4-wide softmax; lanes 16-31 duplicate, writes masked */
  float sc[4]; float mx = -1e30f;
#pragma unroll
  for (int m = 0; m < 4; ++m) {
    float s = 0.0f;
#pragma unroll
    for (int f = 0; f < Fn; ++f)
      s += sQ[wib][(il * 4 + l) * 16 + f] * sK[wib][(il * 4 + m) * 16 + f];
    sc[m] = s * 0.25f;
    mx = fmaxf(mx, sc[m]);
  }
  float den = 0.0f;
#pragma unroll
  for (int m = 0; m < 4; ++m) { sc[m] = expf(sc[m] - mx); den += sc[m]; }
  float inv = 1.0f / den;
#pragma unroll
  for (int f = 0; f < Fn; ++f) {
    float acc = 0.0f;
#pragma unroll
    for (int m = 0; m < 4; ++m) acc += sc[m] * sV[wib][(il * 4 + m) * 16 + f];
    if (half == 0) sC[wib][(il * 4 + l) * 16 + f] = acc * inv;
  }
  __syncthreads();
  for (int o = lane; o < 64; o += 32) {      /* sum over lag dim */
    int oil = o >> 4, f = o & 15;
    float s = 0.0f;
#pragma unroll
    for (int ll = 0; ll < 4; ++ll) s += sC[wib][(oil * 4 + ll) * 16 + f];
    att_out[((i0 + oil) * Vn + v) * Fn + f] = s;
  }
}

/* ---------------- K5: message passing (adj mix) ---------------- */
__global__ void k_msg(const float* __restrict__ att, const float* __restrict__ adj,
                      float* __restrict__ msg) {
  int id = blockIdx.x * blockDim.x + threadIdx.x;
  if (id >= NIn * Vn) return;
  int u = id % Vn;
  int i = id / Vn;
  float acc[Fn];
#pragma unroll
  for (int f = 0; f < Fn; ++f) acc[f] = 0.0f;
  for (int v = 0; v < Vn; ++v) {
    float a = adj[u * Vn + v];
    const float* src = att + (i * Vn + v) * Fn;
#pragma unroll
    for (int f = 0; f < Fn; ++f) acc[f] += a * src[f];
  }
  float* dst = msg + (i * Vn + u) * Fn;
#pragma unroll
  for (int f = 0; f < Fn; ++f) dst[f] = acc[f];
}

/* ------ K6: signal attention (WMMA, wave = 1 window, scrambled reshape) -------- */
__global__ void k_sattn(const float* __restrict__ msg, const float* __restrict__ pe_v,
                        const float* __restrict__ Wq, const float* __restrict__ bq,
                        const float* __restrict__ Wk, const float* __restrict__ bk,
                        const float* __restrict__ Wv, const float* __restrict__ bv,
                        float* __restrict__ s_out) {
  __shared__ float sQ[4][768];
  __shared__ float sK[4][768];
  __shared__ float sV[4][768];
  __shared__ float csum[4][Vn];
  int lane = threadIdx.x & 31;
  int wib  = threadIdx.x >> 5;
  int i    = blockIdx.x * 4 + wib;          /* grid exact: 23552 waves */
  int half = lane >> 4, lr = lane & 15;

  v16h fq, fk, fv;
#pragma unroll
  for (int e = 0; e < 16; ++e) {
    int k = half * 16 + e;
    fq[e] = (k < Fn) ? (_Float16)Wq[k * Fn + lr] : (_Float16)0.0f;
    fk[e] = (k < Fn) ? (_Float16)Wk[k * Fn + lr] : (_Float16)0.0f;
    fv[e] = (k < Fn) ? (_Float16)Wv[k * Fn + lr] : (_Float16)0.0f;
  }
  for (int rt = 0; rt < 3; ++rt) {          /* 34 rows -> 3 tiles of 16 */
    int n = rt * 16 + lr;
    v16h a;
#pragma unroll
    for (int e = 0; e < 16; ++e) {
      int k = a_kmap(e, half);
      float val = 0.0f;
      if (k < Fn && n < Vn) {
        /* bug-faithful raw reshape [B,V,W,F] -> [B*W,V,F] */
        int g  = i * (Vn * Fn) + n * Fn + k;
        int b  = g / (Vn * Wn * Fn);
        int r  = g - b * (Vn * Wn * Fn);
        int vs = r / (Wn * Fn);
        int r2 = r - vs * (Wn * Fn);
        int w  = r2 >> 4;
        val = msg[((b * Wn + w) * Vn + vs) * Fn + k] + pe_v[vs * Fn + k];
      }
      a[e] = (_Float16)val;
    }
    v8f z = {};
    v8f cq = wmma16(a, fq, z);
    v8f ck = wmma16(a, fk, z);
    v8f cv = wmma16(a, fv, z);
#pragma unroll
    for (int j = 0; j < 8; ++j) {
      int r = rt * 16 + j + 8 * half;
      sQ[wib][r * 16 + lr] = cq[j] + bq[lr];
      sK[wib][r * 16 + lr] = ck[j] + bk[lr];
      sV[wib][r * 16 + lr] = cv[j] + bv[lr];
    }
  }
  for (int m = lane; m < Vn; m += 32) csum[wib][m] = 0.0f;
  __syncthreads();
  for (int n = lane; n < Vn; n += 32) {
    float sb[Vn]; float mx = -1e30f;
#pragma unroll
    for (int m = 0; m < Vn; ++m) {
      float s = 0.0f;
#pragma unroll
      for (int f = 0; f < Fn; ++f) s += sQ[wib][n * 16 + f] * sK[wib][m * 16 + f];
      sb[m] = s * 0.25f; mx = fmaxf(mx, sb[m]);
    }
    float den = 0.0f;
#pragma unroll
    for (int m = 0; m < Vn; ++m) { sb[m] = expf(sb[m] - mx); den += sb[m]; }
    float inv = 1.0f / den;
#pragma unroll
    for (int m = 0; m < Vn; ++m) atomicAdd(&csum[wib][m], sb[m] * inv);
  }
  __syncthreads();
  for (int d = lane; d < Fn; d += 32) {
    float acc = 0.0f;
#pragma unroll
    for (int m = 0; m < Vn; ++m) acc += csum[wib][m] * sV[wib][m * 16 + d];
    s_out[i * Fn + d] = acc;
  }
}

/* --------- K7: generic WMMA GEMM: C = A[M,K]@B[K,N] + bias (+opt PE) ---------- */
__global__ void k_gemm_wmma(const float* __restrict__ A, int lda,
                            const float* __restrict__ Bw, int ldb,
                            const float* __restrict__ bias,
                            const float* __restrict__ pe, int pe_period,
                            float* __restrict__ C, int ldc,
                            int Mdim, int Ndim, int Kdim) {
  int lane = threadIdx.x & 31;
  int wid  = blockIdx.x * (blockDim.x >> 5) + (threadIdx.x >> 5);
  int ntn = (Ndim + 15) >> 4;
  int mt = wid / ntn, nt = wid % ntn;
  if (mt * 16 >= Mdim) return;
  int half = lane >> 4, lr = lane & 15;
  int row  = mt * 16 + lr;
  int colb = nt * 16 + lr;
  v8f acc = {};
  for (int kk = 0; kk < Kdim; kk += 32) {
    v16h a, b;
#pragma unroll
    for (int e = 0; e < 16; ++e) {
      int k = kk + a_kmap(e, half);
      float av = 0.0f;
      if (k < Kdim && row < Mdim) {
        av = A[row * lda + k];
        if (pe) av += pe[(row % pe_period) * Kdim + k];
      }
      a[e] = (_Float16)av;
      int kb = kk + half * 16 + e;
      float bv = (kb < Kdim && colb < Ndim) ? Bw[kb * ldb + colb] : 0.0f;
      b[e] = (_Float16)bv;
    }
    acc = wmma16(a, b, acc);
  }
#pragma unroll
  for (int j = 0; j < 8; ++j) {
    int r = mt * 16 + j + 8 * half;
    if (r < Mdim && colb < Ndim)
      C[r * ldc + colb] = acc[j] + (bias ? bias[colb] : 0.0f);
  }
}

/* --- K8: GRU layer (WMMA; block = 16 batch rows, 4 waves = 4 col-groups) ------- */
__global__ void k_gru(const float* __restrict__ xW, const float* __restrict__ Whh,
                      const float* __restrict__ bhh, float* __restrict__ ys) {
  __shared__ float hbuf[16 * Hn];
  int lane = threadIdx.x & 31;
  int ct   = threadIdx.x >> 5;               /* wave = column group 0..3 */
  int rt   = blockIdx.x;                     /* 32 tiles of 16 batch rows */
  int half = lane >> 4, lr = lane & 15;
  for (int o = threadIdx.x; o < 16 * Hn; o += 128) hbuf[o] = 0.0f;
  __syncthreads();

  /* Whh fragments for this wave's gate columns: invariant over time steps */
  v16h br[2], bz[2], bn[2];
#pragma unroll
  for (int kb = 0; kb < 2; ++kb)
#pragma unroll
    for (int e = 0; e < 16; ++e) {
      int k = kb * 32 + half * 16 + e;
      br[kb][e] = (_Float16)Whh[k * H3 + ct * 16 + lr];
      bz[kb][e] = (_Float16)Whh[k * H3 + Hn + ct * 16 + lr];
      bn[kb][e] = (_Float16)Whh[k * H3 + 2 * Hn + ct * 16 + lr];
    }

  for (int t = 0; t < Wn; ++t) {
    v8f dr = {}, dz = {}, dn = {};
#pragma unroll
    for (int kb = 0; kb < 2; ++kb) {
      v16h a;
#pragma unroll
      for (int e = 0; e < 16; ++e)
        a[e] = (_Float16)hbuf[lr * Hn + kb * 32 + a_kmap(e, half)];
      dr = wmma16(a, br[kb], dr);
      dz = wmma16(a, bz[kb], dz);
      dn = wmma16(a, bn[kb], dn);
    }
    float hnew[8];
#pragma unroll
    for (int j = 0; j < 8; ++j) {
      int row = j + 8 * half;
      int c = ct * 16 + lr;
      int bg = rt * 16 + row;
      const float* xw = xW + (bg * Wn + t) * H3;
      if (t + 1 < Wn) __builtin_prefetch(xw + H3, 0, 1);   /* next step's inputs */
      float r = 1.0f / (1.0f + expf(-(xw[c] + dr[j] + bhh[c])));
      float z = 1.0f / (1.0f + expf(-(xw[Hn + c] + dz[j] + bhh[Hn + c])));
      float n = tanhf(xw[2 * Hn + c] + r * (dn[j] + bhh[2 * Hn + c]));
      float hold = hbuf[row * Hn + c];
      hnew[j] = (1.0f - z) * n + z * hold;
    }
    __syncthreads();                         /* all reads of h(t) done */
#pragma unroll
    for (int j = 0; j < 8; ++j) {
      int row = j + 8 * half;
      int c = ct * 16 + lr;
      hbuf[row * Hn + c] = hnew[j];
      ys[((rt * 16 + row) * Wn + t) * Hn + c] = hnew[j];
    }
    __syncthreads();                         /* h(t+1) fully published */
  }
}

/* ---------------- K9: temporal attention pooling ---------------- */
__global__ void k_tattn(const float* __restrict__ Qt, const float* __restrict__ Kt,
                        const float* __restrict__ Vt, float* __restrict__ tval) {
  __shared__ float csum[8][Wn];
  int lane = threadIdx.x & 31;
  int wib = threadIdx.x >> 5;
  int b = blockIdx.x * 8 + wib;             /* grid exact: 512 waves */
  for (int m = lane; m < Wn; m += 32) csum[wib][m] = 0.0f;
  __syncthreads();
  for (int n = lane; n < Wn; n += 32) {
    float sb[Wn]; float mx = -1e30f;
    const float* qr = Qt + (b * Wn + n) * Hn;
    for (int m = 0; m < Wn; ++m) {
      const float* kr = Kt + (b * Wn + m) * Hn;
      __builtin_prefetch(Vt + (b * Wn + m) * Hn, 0, 1);  /* warm Vt for phase 3 */
      float s = 0.0f;
#pragma unroll 8
      for (int d = 0; d < Hn; ++d) s += qr[d] * kr[d];
      sb[m] = s * 0.125f; mx = fmaxf(mx, sb[m]);
    }
    float den = 0.0f;
    for (int m = 0; m < Wn; ++m) { sb[m] = expf(sb[m] - mx); den += sb[m]; }
    float inv = 1.0f / den;
    for (int m = 0; m < Wn; ++m) atomicAdd(&csum[wib][m], sb[m] * inv);
  }
  __syncthreads();
  for (int d = lane; d < Hn; d += 32) {
    float acc = 0.0f;
    for (int m = 0; m < Wn; ++m) acc += csum[wib][m] * Vt[(b * Wn + m) * Hn + d];
    tval[b * Hn + d] = acc;
  }
}

/* ---------------- K10: classifier ---------------- */
__global__ void k_cls(const float* __restrict__ tval,
                      const float* __restrict__ W1, const float* __restrict__ b1,
                      const float* __restrict__ W2, const float* __restrict__ b2,
                      float* __restrict__ out) {
  int b = blockIdx.x * blockDim.x + threadIdx.x;
  if (b >= Bn) return;
  float h[Hn];
  for (int j = 0; j < Hn; ++j) {
    float acc = b1[j];
#pragma unroll 8
    for (int k = 0; k < Hn; ++k) acc += tval[b * Hn + k] * W1[k * Hn + j];
    h[j] = fmaxf(acc, 0.0f);
  }
  float o = b2[0];
  for (int j = 0; j < Hn; ++j) o += h[j] * W2[j];
  out[b] = o;
}

extern "C" void kernel_launch(void* const* d_in, const int* in_sizes, int n_in,
                              void* d_out, int out_size, void* d_ws, size_t ws_size,
                              hipStream_t stream) {
  const float* arr    = (const float*)d_in[0];
  const float* mask   = (const float*)d_in[1];
  const float* tim    = (const float*)d_in[2];
  const float* wdec   = (const float*)d_in[3];
  const float* bdec   = (const float*)d_in[4];
  const float* enc_W1 = (const float*)d_in[5];
  const float* enc_b1 = (const float*)d_in[6];
  const float* enc_W2 = (const float*)d_in[7];
  const float* enc_b2 = (const float*)d_in[8];
  const float* tla_Wq = (const float*)d_in[9];
  const float* tla_bq = (const float*)d_in[10];
  const float* tla_Wk = (const float*)d_in[11];
  const float* tla_bk = (const float*)d_in[12];
  const float* tla_Wv = (const float*)d_in[13];
  const float* tla_bv = (const float*)d_in[14];
  const float* adj    = (const float*)d_in[15];
  /* d_in[16..19] = mp_* : dead code in reference */
  const float* sa_Wq  = (const float*)d_in[20];
  const float* sa_bq  = (const float*)d_in[21];
  const float* sa_Wk  = (const float*)d_in[22];
  const float* sa_bk  = (const float*)d_in[23];
  const float* sa_Wv  = (const float*)d_in[24];
  const float* sa_bv  = (const float*)d_in[25];
  const float* Wih0   = (const float*)d_in[26];
  const float* Whh0   = (const float*)d_in[27];
  const float* bih0   = (const float*)d_in[28];
  const float* bhh0   = (const float*)d_in[29];
  const float* Wih1   = (const float*)d_in[30];
  const float* Whh1   = (const float*)d_in[31];
  const float* bih1   = (const float*)d_in[32];
  const float* bhh1   = (const float*)d_in[33];
  const float* ta_Wq  = (const float*)d_in[34];
  const float* ta_bq  = (const float*)d_in[35];
  const float* ta_Wk  = (const float*)d_in[36];
  const float* ta_bk  = (const float*)d_in[37];
  const float* ta_Wv  = (const float*)d_in[38];
  const float* ta_bv  = (const float*)d_in[39];
  const float* cls_W1 = (const float*)d_in[40];
  const float* cls_b1 = (const float*)d_in[41];
  const float* cls_W2 = (const float*)d_in[42];
  const float* cls_b2 = (const float*)d_in[43];
  (void)in_sizes; (void)n_in; (void)out_size; (void)ws_size;

  float* ws      = (float*)d_ws;
  float* pe_win  = ws + OFF_PEWIN;
  float* pe_v    = ws + OFF_PEV;
  float* pe_w    = ws + OFF_PEW;
  unsigned* cmax = (unsigned*)(ws + OFF_CMAX);
  float* cum     = ws + OFF_CUM;
  float* last    = ws + OFF_LAST;
  float* meanv   = ws + OFF_MEAN;
  float* ximp    = ws + OFF_XIMP;
  float* s_out   = ws + OFF_SOUT;
  float* tval    = ws + OFF_TVAL;
  /* region A: enc, then message, then Qt/Kt/Vt */
  float* enc     = ws + OFF_A;
  float* message = ws + OFF_A;
  float* Qt      = ws + OFF_A;
  float* Kt      = ws + OFF_A + NIn * Hn;
  float* Vt      = ws + OFF_A + 2 * NIn * Hn;
  /* region B: att_out, then xW0|g1|xW1|g2 */
  float* att_out = ws + OFF_B;
  float* xW0     = ws + OFF_B;
  float* g1      = ws + OFF_B + NIn * H3;
  float* xW1     = ws + OFF_B + NIn * H3 + NIn * Hn;
  float* g2      = ws + OFF_B + 2 * NIn * H3 + NIn * Hn;

  k_init<<<12, 256, 0, stream>>>(pe_win, pe_v, pe_w, cmax);
  k_scan<<<(Bn * Vn) / 256, 256, 0, stream>>>(arr, mask, tim, cum, last, meanv, cmax);
  k_apply<<<(Bn * Tn * Vn) / 256, 256, 0, stream>>>(arr, mask, cum, last, meanv, cmax,
                                                    wdec, bdec, ximp);
  k_enc<<<(Bn * Tn * Vn) / 256, 256, 0, stream>>>(ximp, mask, enc_W1, enc_b1,
                                                  enc_W2, enc_b2, enc);
  k_tla<<<(Vn * (NIn / 4)) / 8, 256, 0, stream>>>(enc, pe_win, tla_Wq, tla_bq,
                                                  tla_Wk, tla_bk, tla_Wv, tla_bv, att_out);
  k_msg<<<(NIn * Vn) / 256, 256, 0, stream>>>(att_out, adj, message);
  k_sattn<<<NIn / 4, 128, 0, stream>>>(message, pe_v, sa_Wq, sa_bq, sa_Wk, sa_bk,
                                       sa_Wv, sa_bv, s_out);
  /* GRU layer 0 */
  k_gemm_wmma<<<((NIn / 16) * (H3 / 16)) / 8, 256, 0, stream>>>(
      s_out, Fn, Wih0, H3, bih0, (const float*)nullptr, 1, xW0, H3, NIn, H3, Fn);
  k_gru<<<32, 128, 0, stream>>>(xW0, Whh0, bhh0, g1);
  /* GRU layer 1 */
  k_gemm_wmma<<<((NIn / 16) * (H3 / 16)) / 8, 256, 0, stream>>>(
      g1, Hn, Wih1, H3, bih1, (const float*)nullptr, 1, xW1, H3, NIn, H3, Hn);
  k_gru<<<32, 128, 0, stream>>>(xW1, Whh1, bhh1, g2);
  /* temporal attention projections: xt = g2 + pe_w[l] */
  k_gemm_wmma<<<((NIn / 16) * (Hn / 16)) / 8, 256, 0, stream>>>(
      g2, Hn, ta_Wq, Hn, ta_bq, pe_w, Wn, Qt, Hn, NIn, Hn, Hn);
  k_gemm_wmma<<<((NIn / 16) * (Hn / 16)) / 8, 256, 0, stream>>>(
      g2, Hn, ta_Wk, Hn, ta_bk, pe_w, Wn, Kt, Hn, NIn, Hn, Hn);
  k_gemm_wmma<<<((NIn / 16) * (Hn / 16)) / 8, 256, 0, stream>>>(
      g2, Hn, ta_Wv, Hn, ta_bv, pe_w, Wn, Vt, Hn, NIn, Hn, Hn);
  k_tattn<<<Bn / 8, 256, 0, stream>>>(Qt, Kt, Vt, tval);
  k_cls<<<2, 256, 0, stream>>>(tval, cls_W1, cls_b1, cls_W2, cls_b2, (float*)d_out);
}
